// LSTMAttnDecoder_37220186587600
// MI455X (gfx1250) — matrix operand order repaired and checked by reading, not compiled, not run on hardware
//
#include <hip/hip_runtime.h>
#include <hip/hip_bf16.h>
#include <math.h>

#define B_  64
#define L_  400
#define E_  512
#define H_  1024
#define V_  32000

typedef __attribute__((ext_vector_type(16))) __bf16 v16bf;
typedef __attribute__((ext_vector_type(8)))  __bf16 v8bf;
typedef __attribute__((ext_vector_type(8)))  float  v8f;

// round-to-nearest-even f32 -> bf16
__device__ __forceinline__ __bf16 f2bf(float f) {
  union { float f; unsigned u; } x; x.f = f;
  unsigned u = x.u + 0x7FFFu + ((x.u >> 16) & 1u);
  unsigned short h = (unsigned short)(u >> 16);
  __bf16 r; __builtin_memcpy(&r, &h, 2);
  return r;
}
// two f32 -> packed bf16x2 (RNE)
__device__ __forceinline__ unsigned pack_bf2(float a, float b) {
  union { float f; unsigned u; } x, y; x.f = a; y.f = b;
  unsigned ua = x.u + 0x7FFFu + ((x.u >> 16) & 1u);
  unsigned ub = y.u + 0x7FFFu + ((y.u >> 16) & 1u);
  return (ua >> 16) | (ub & 0xFFFF0000u);
}

// ---------------------------------------------------------------------------
// Generic BF16 WMMA GEMM:  C[M,N] (+=) A[M,K](f32,row-major) * B (+bias0+bias1)
// B_IS_NK==true : B given as N x K (computes A @ B^T, e.g. W_ih (4H,E))
// B_IS_NK==false: B given as K x N (e.g. W_enc (2H,H))
// 256 threads = 8 wave32s; BK=32 (one v_wmma_f32_16x16x32_bf16 K-step).
// LDS: As[r][k], Bs[n][k]; fragments are contiguous b128 DS loads (ISA 7.12.2).
// Fast path (full tile in range): float4 global loads + packed b64 LDS stores.
// ---------------------------------------------------------------------------
template<int BM, int BN, bool B_IS_NK, bool ACC>
__global__ __launch_bounds__(256) void wmma_gemm(
    const float* __restrict__ A, int lda,
    const float* __restrict__ Bg, int ldb,
    float* __restrict__ C, int ldc,
    const float* __restrict__ bias0, const float* __restrict__ bias1,
    int M, int N, int K)
{
  constexpr int BK = 32;
  constexpr int LD = BK + 8;            // 40 elems = 80B rows (16B aligned)
  constexpr int WROWS = BM / 32;        // waves along M
  constexpr int WCOLS = 8 / WROWS;      // waves along N
  constexpr int TN = BN / (WCOLS * 16); // 16-wide frags per wave in N
  static_assert(BM % 32 == 0 && (8 % WROWS) == 0 && TN >= 1, "tile cfg");

  __shared__ __bf16 As[BM * LD];
  __shared__ __bf16 Bs[BN * LD];

  const int tid  = threadIdx.x;
  const int lane = tid & 31;
  const int wave = tid >> 5;
  const int wm   = wave % WROWS;
  const int wn   = wave / WROWS;
  const int bm   = blockIdx.y * BM;
  const int bn   = blockIdx.x * BN;
  const int half = lane >> 4;
  const int lr   = lane & 15;

  v8f acc[2][TN] = {};

  for (int k0 = 0; k0 < K; k0 += BK) {
    const bool fastA = (bm + BM <= M) && (k0 + BK <= K);
    const bool fastB = (bn + BN <= N) && (k0 + BK <= K);

    // ---- stage A tile (f32 -> bf16) ----
    if (fastA) {
      #pragma unroll
      for (int it = 0; it < (BM * BK) / 1024; it++) {
        int idx = it * 256 + tid;            // one float4 per idx
        int r  = idx >> 3;                   // / (BK/4)
        int kq = (idx & 7) << 2;
        const float4 v = *(const float4*)&A[(size_t)(bm + r) * lda + (k0 + kq)];
        uint2 p; p.x = pack_bf2(v.x, v.y); p.y = pack_bf2(v.z, v.w);
        *(uint2*)&As[r * LD + kq] = p;
      }
      if (k0 + BK < K) {
        int r = bm + (tid % BM);
        __builtin_prefetch(&A[(size_t)r * lda + k0 + BK], 0, 1);
      }
    } else {
      for (int idx = tid; idx < BM * BK; idx += 256) {
        int r = idx / BK, k = idx % BK;
        int gm = bm + r, gk = k0 + k;
        float v = 0.f;
        if (gm < M && gk < K) v = A[(size_t)gm * lda + gk];
        As[r * LD + k] = f2bf(v);
      }
    }
    // ---- stage B tile as [n][k] ----
    if (fastB) {
      if (B_IS_NK) {
        #pragma unroll
        for (int it = 0; it < (BN * BK) / 1024; it++) {
          int idx = it * 256 + tid;
          int n  = idx >> 3;
          int kq = (idx & 7) << 2;
          const float4 v = *(const float4*)&Bg[(size_t)(bn + n) * ldb + (k0 + kq)];
          uint2 p; p.x = pack_bf2(v.x, v.y); p.y = pack_bf2(v.z, v.w);
          *(uint2*)&Bs[n * LD + kq] = p;
        }
      } else {
        #pragma unroll
        for (int it = 0; it < (BN * 8) / 256; it++) {
          int idx = it * 256 + tid;          // BN*8 quads total
          int k4 = idx / BN;                 // 4 consecutive k for one n
          int n  = idx % BN;                 // consecutive tid -> consecutive n
          const float* g = &Bg[(size_t)(k0 + k4 * 4) * ldb + (bn + n)];
          float a0 = g[0], a1 = g[(size_t)ldb], a2 = g[(size_t)2 * ldb], a3 = g[(size_t)3 * ldb];
          uint2 p; p.x = pack_bf2(a0, a1); p.y = pack_bf2(a2, a3);
          *(uint2*)&Bs[n * LD + k4 * 4] = p;
        }
      }
    } else {
      for (int idx = tid; idx < BN * BK; idx += 256) {
        int n = idx % BN, k = idx / BN;
        int gn = bn + n, gk = k0 + k;
        float v = 0.f;
        if (gn < N && gk < K) v = B_IS_NK ? Bg[(size_t)gn * ldb + gk]
                                          : Bg[(size_t)gk * ldb + gn];
        Bs[n * LD + k] = f2bf(v);
      }
    }
    __syncthreads();

    // ---- A fragments: lane holds M=lr, K = half*8 + [0..7] and +16 ----
    v16bf af[2];
    #pragma unroll
    for (int tm = 0; tm < 2; tm++) {
      const __bf16* p = &As[(wm * 32 + tm * 16 + lr) * LD + half * 8];
      v8bf lo = *(const v8bf*)p;
      v8bf hi = *(const v8bf*)(p + 16);
      af[tm] = __builtin_shufflevector(lo, hi, 0,1,2,3,4,5,6,7,8,9,10,11,12,13,14,15);
    }
    // ---- B fragments + WMMA: lane holds N=lr, K = half*8 + [0..7] and +16 ----
    #pragma unroll
    for (int tn = 0; tn < TN; tn++) {
      const __bf16* p = &Bs[(wn * (TN * 16) + tn * 16 + lr) * LD + half * 8];
      v8bf lo = *(const v8bf*)p;
      v8bf hi = *(const v8bf*)(p + 16);
      v16bf bfrag = __builtin_shufflevector(lo, hi, 0,1,2,3,4,5,6,7,8,9,10,11,12,13,14,15);
      #pragma unroll
      for (int tm = 0; tm < 2; tm++) {
        acc[tm][tn] = __builtin_amdgcn_wmma_f32_16x16x32_bf16(
            false, af[tm], false, bfrag, (short)0, acc[tm][tn], false, false);
      }
    }
    __syncthreads();
  }

  // ---- store: VGPR j -> M = 16*tm + half*8 + j, N = lr ----
  #pragma unroll
  for (int tm = 0; tm < 2; tm++) {
    #pragma unroll
    for (int tn = 0; tn < TN; tn++) {
      int gn = bn + wn * (TN * 16) + tn * 16 + lr;
      #pragma unroll
      for (int j = 0; j < 8; j++) {
        int gm = bm + wm * 32 + tm * 16 + half * 8 + j;
        if (gm < M && gn < N) {
          float v = acc[tm][tn][j];
          if (bias0) v += bias0[gn];
          if (bias1) v += bias1[gn];
          size_t ci = (size_t)gm * ldc + gn;
          if (ACC) v += C[ci];
          C[ci] = v;
        }
      }
    }
  }
}

// ---------------------------------------------------------------------------
// Small elementwise / reduction kernels
// ---------------------------------------------------------------------------
__global__ void k_embed_cov(const int* __restrict__ word, const float* __restrict__ emb,
                            const float* __restrict__ attw, const float* __restrict__ covv,
                            float* __restrict__ xb, float* __restrict__ cov_out) {
  int idx = blockIdx.x * 256 + threadIdx.x;
  if (idx < B_ * E_) {
    int b = idx / E_, e = idx % E_;
    xb[idx] = emb[(size_t)word[b] * E_ + e];
  }
  if (idx < B_ * L_) cov_out[idx] = covv[idx] + attw[idx];
}

__global__ void k_lstm(const float* __restrict__ gates, const float* __restrict__ c0,
                       float* __restrict__ s_t, float* __restrict__ c_t) {
  int idx = blockIdx.x * 256 + threadIdx.x;
  if (idx >= B_ * H_) return;
  int b = idx / H_, h = idx % H_;
  const float* g = gates + (size_t)b * 4 * H_;
  float gi = g[h], gf = g[H_ + h], gg = g[2 * H_ + h], go = g[3 * H_ + h];
  float si = 1.f / (1.f + expf(-gi));
  float sf = 1.f / (1.f + expf(-gf));
  float so = 1.f / (1.f + expf(-go));
  float c = sf * c0[idx] + si * tanhf(gg);
  c_t[idx] = c;
  s_t[idx] = so * tanhf(c);
}

__global__ void k_mean(const float* __restrict__ enc_proj, float* __restrict__ meanb) {
  int b = blockIdx.x;
  int h = blockIdx.y * 256 + threadIdx.x;
  const float* p = enc_proj + (size_t)b * L_ * H_ + h;
  float s = 0.f;
  for (int l = 0; l < L_; l++) s += p[(size_t)l * H_];
  meanb[b * H_ + h] = s * (1.0f / L_);
}

__global__ void k_feat(const float* __restrict__ meanb, const float* __restrict__ decp,
                       const float* __restrict__ covp, float* __restrict__ feat) {
  int idx = blockIdx.x * 256 + threadIdx.x;
  if (idx < B_ * H_) feat[idx] = tanhf(meanb[idx] + decp[idx] + covp[idx]);
}

__global__ __launch_bounds__(512) void k_softmax_L(const float* __restrict__ e,
                                                   float* __restrict__ a_t) {
  __shared__ float red[512];
  int b = blockIdx.x, t = threadIdx.x;
  float v = (t < L_) ? e[b * L_ + t] : -3.4e38f;
  red[t] = v; __syncthreads();
  for (int s = 256; s > 0; s >>= 1) { if (t < s) red[t] = fmaxf(red[t], red[t + s]); __syncthreads(); }
  float mx = red[0]; __syncthreads();
  float ex = (t < L_) ? expf(v - mx) : 0.f;
  red[t] = ex; __syncthreads();
  for (int s = 256; s > 0; s >>= 1) { if (t < s) red[t] += red[t + s]; __syncthreads(); }
  if (t < L_) a_t[b * L_ + t] = ex / red[0];
}

// context[b,h] = sum_l a_t[b,l] * enc_proj[b,l,h]; also builds [s_t, ctx] concat.
// Stages the 400 attention weights into LDS with async global->LDS copies.
__global__ void k_context(const float* __restrict__ a_t, const float* __restrict__ enc_proj,
                          const float* __restrict__ s_t, float* __restrict__ ctx,
                          float* __restrict__ pv_in) {
  __shared__ float a_s[L_];                 // 400 floats = 100 x b128
  int b = blockIdx.x;
  int t = threadIdx.x;
  if (t < L_ / 4) {
    unsigned lds_off = (unsigned)(size_t)&a_s[t * 4];
    const float* g = a_t + (size_t)b * L_ + t * 4;
    asm volatile("global_load_async_to_lds_b128 %0, %1, off"
                 :: "v"(lds_off), "v"(g) : "memory");
  }
  asm volatile("s_wait_asynccnt 0x0" ::: "memory");
  __syncthreads();

  int h = blockIdx.y * 256 + t;
  const float* p = enc_proj + (size_t)b * L_ * H_ + h;
  float s = 0.f;
  for (int l = 0; l < L_; l++) s += a_s[l] * p[(size_t)l * H_];
  ctx[b * H_ + h] = s;
  pv_in[(size_t)b * 2 * H_ + H_ + h] = s;   // [s_t, context] concat
  pv_in[(size_t)b * 2 * H_ + h]      = s_t[b * H_ + h];
}

__global__ __launch_bounds__(256) void k_pgen(const float* __restrict__ ctx,
    const float* __restrict__ s_t, const float* __restrict__ xb,
    const float* __restrict__ w_ctx, const float* __restrict__ w_st,
    const float* __restrict__ w_x, const float* __restrict__ b_ctx,
    const float* __restrict__ b_st, const float* __restrict__ b_x,
    float* __restrict__ pgen) {
  __shared__ float red[256];
  int b = blockIdx.x, t = threadIdx.x;
  float s = 0.f;
  for (int h = t; h < H_; h += 256) s += ctx[b * H_ + h] * w_ctx[h] + s_t[b * H_ + h] * w_st[h];
  for (int e = t; e < E_; e += 256) s += xb[b * E_ + e] * w_x[e];
  red[t] = s; __syncthreads();
  for (int st = 128; st > 0; st >>= 1) { if (t < st) red[t] += red[t + st]; __syncthreads(); }
  if (t == 0) {
    float z = red[0] + b_ctx[0] + b_st[0] + b_x[0];
    pgen[b] = 1.f / (1.f + expf(-z));
  }
}

__global__ void k_zero(float* __restrict__ p, int n) {
  int i = blockIdx.x * 256 + threadIdx.x;
  if (i < n) p[i] = 0.f;
}

__global__ void k_scatter(const int* __restrict__ src_ids, const float* __restrict__ a_t,
                          float* __restrict__ copyd) {
  int idx = blockIdx.x * 256 + threadIdx.x;
  if (idx >= B_ * L_) return;
  int b = idx / L_;
  atomicAdd(&copyd[(size_t)b * V_ + src_ids[idx]], a_t[idx]);
}

__global__ __launch_bounds__(1024) void k_softmax_V_stats(const float* __restrict__ logits,
    float* __restrict__ vmax, float* __restrict__ vsum) {
  __shared__ float red[1024];
  int b = blockIdx.x, t = threadIdx.x;
  const float* p = logits + (size_t)b * V_;
  float mx = -3.4e38f;
  for (int v = t; v < V_; v += 1024) mx = fmaxf(mx, p[v]);
  red[t] = mx; __syncthreads();
  for (int s = 512; s > 0; s >>= 1) { if (t < s) red[t] = fmaxf(red[t], red[t + s]); __syncthreads(); }
  mx = red[0]; __syncthreads();
  float sm = 0.f;
  for (int v = t; v < V_; v += 1024) sm += expf(p[v] - mx);
  red[t] = sm; __syncthreads();
  for (int s = 512; s > 0; s >>= 1) { if (t < s) red[t] += red[t + s]; __syncthreads(); }
  if (t == 0) { vmax[b] = mx; vsum[b] = red[0]; }
}

__global__ void k_final(const float* __restrict__ logits, const float* __restrict__ copyd,
                        const float* __restrict__ pgen, const float* __restrict__ vmax,
                        const float* __restrict__ vsum, float* __restrict__ P) {
  size_t idx = (size_t)blockIdx.x * 256 + threadIdx.x;
  if (idx >= (size_t)B_ * V_) return;
  int b = (int)(idx / V_);
  float pg = pgen[b];
  float pv = expf(logits[idx] - vmax[b]) / vsum[b];
  P[idx] = pg * pv + (1.f - pg) * copyd[idx];
}

// ---------------------------------------------------------------------------
extern "C" void kernel_launch(void* const* d_in, const int* in_sizes, int n_in,
                              void* d_out, int out_size, void* d_ws, size_t ws_size,
                              hipStream_t stream) {
  (void)in_sizes; (void)n_in; (void)out_size; (void)ws_size;
  const int*   word  = (const int*)  d_in[0];
  const float* h0    = (const float*)d_in[1];
  const float* c0    = (const float*)d_in[2];
  const float* enc   = (const float*)d_in[3];
  const float* attw  = (const float*)d_in[4];
  const float* covv  = (const float*)d_in[5];
  const int*   src   = (const int*)  d_in[6];
  const float* emb   = (const float*)d_in[7];
  const float* W_ih  = (const float*)d_in[8];
  const float* W_hh  = (const float*)d_in[9];
  const float* b_ih  = (const float*)d_in[10];
  const float* b_hh  = (const float*)d_in[11];
  const float* W_enc = (const float*)d_in[12];
  const float* b_enc = (const float*)d_in[13];
  const float* W_dec = (const float*)d_in[14];
  const float* b_dec = (const float*)d_in[15];
  const float* W_cov = (const float*)d_in[16];
  const float* b_cov = (const float*)d_in[17];
  const float* W_e   = (const float*)d_in[18];
  const float* b_e   = (const float*)d_in[19];
  const float* W_pv1 = (const float*)d_in[20];
  const float* b_pv1 = (const float*)d_in[21];
  const float* W_pv2 = (const float*)d_in[22];
  const float* b_pv2 = (const float*)d_in[23];
  const float* w_ctx = (const float*)d_in[24];
  const float* b_ctx = (const float*)d_in[25];
  const float* w_st  = (const float*)d_in[26];
  const float* b_st  = (const float*)d_in[27];
  const float* w_x   = (const float*)d_in[28];
  const float* b_x   = (const float*)d_in[29];

  // outputs: (P_final, s_t, c, a_t, coverage)
  float* out = (float*)d_out;
  float* P_final = out;
  float* s_t  = P_final + (size_t)B_ * V_;
  float* c_t  = s_t + B_ * H_;
  float* a_t  = c_t + B_ * H_;
  float* covo = a_t + B_ * L_;

  // workspace
  float* w = (float*)d_ws;
  float* enc_proj = w; w += (size_t)B_ * L_ * H_;
  float* xb     = w; w += B_ * E_;
  float* gates  = w; w += (size_t)B_ * 4 * H_;
  float* meanb  = w; w += B_ * H_;
  float* decp   = w; w += B_ * H_;
  float* covp   = w; w += B_ * H_;
  float* feat   = w; w += B_ * H_;
  float* eb     = w; w += B_ * L_;
  float* ctx    = w; w += B_ * H_;
  float* pv_in  = w; w += (size_t)B_ * 2 * H_;
  float* pv_h   = w; w += B_ * H_;
  float* logits = w; w += (size_t)B_ * V_;
  float* copyd  = w; w += (size_t)B_ * V_;
  float* pgen   = w; w += B_;
  float* vmax   = w; w += B_;
  float* vsum   = w; w += B_;

  // 1. embedding gather + coverage
  k_embed_cov<<<(B_ * E_ + 255) / 256, 256, 0, stream>>>(word, emb, attw, covv, xb, covo);
  // zero copy-dist accumulator
  k_zero<<<(B_ * V_ + 255) / 256, 256, 0, stream>>>(copyd, B_ * V_);
  // 2. big GEMM: enc_proj[25600,1024] = encoder_outputs[25600,2048] @ W_enc
  //    BN=512 so the 210MB A stream is read only twice; W_enc stays L2-resident.
  wmma_gemm<32, 512, false, false><<<dim3(H_ / 512, (B_ * L_) / 32), 256, 0, stream>>>(
      enc, 2 * H_, W_enc, H_, enc_proj, H_, b_enc, nullptr, B_ * L_, H_, 2 * H_);
  // 3. LSTM gates: x @ W_ih^T + b_ih + b_hh, then += h0 @ W_hh^T
  wmma_gemm<64, 64, true, false><<<dim3(4 * H_ / 64, 1), 256, 0, stream>>>(
      xb, E_, W_ih, E_, gates, 4 * H_, b_ih, b_hh, B_, 4 * H_, E_);
  wmma_gemm<64, 64, true, true><<<dim3(4 * H_ / 64, 1), 256, 0, stream>>>(
      h0, H_, W_hh, H_, gates, 4 * H_, nullptr, nullptr, B_, 4 * H_, H_);
  // 4. LSTM cell -> s_t, c
  k_lstm<<<(B_ * H_ + 255) / 256, 256, 0, stream>>>(gates, c0, s_t, c_t);
  // 5. mean over L of enc_proj
  k_mean<<<dim3(B_, H_ / 256), 256, 0, stream>>>(enc_proj, meanb);
  // 6. dec_proj = s_t @ W_dec + b_dec
  wmma_gemm<64, 64, false, false><<<dim3(H_ / 64, 1), 256, 0, stream>>>(
      s_t, H_, W_dec, H_, decp, H_, b_dec, nullptr, B_, H_, H_);
  // 7. cov_proj = coverage @ W_cov + b_cov   (K=400 -> edge path on last K-tile)
  wmma_gemm<64, 64, false, false><<<dim3(H_ / 64, 1), 256, 0, stream>>>(
      covo, L_, W_cov, H_, covp, H_, b_cov, nullptr, B_, H_, L_);
  // 8. feat = tanh(mean + dec + cov)
  k_feat<<<(B_ * H_ + 255) / 256, 256, 0, stream>>>(meanb, decp, covp, feat);
  // 9. e = feat @ W_e + b_e   (N=400, edge-guarded)
  wmma_gemm<64, 64, false, false><<<dim3((L_ + 63) / 64, 1), 256, 0, stream>>>(
      feat, H_, W_e, L_, eb, L_, b_e, nullptr, B_, L_, H_);
  // 10. a_t = softmax(e)
  k_softmax_L<<<B_, 512, 0, stream>>>(eb, a_t);
  // 11. context + concat [s_t, context]  (async global->LDS staging of a_t)
  k_context<<<dim3(B_, H_ / 256), 256, 0, stream>>>(a_t, enc_proj, s_t, ctx, pv_in);
  // 12. p_gen
  k_pgen<<<B_, 256, 0, stream>>>(ctx, s_t, xb, w_ctx, w_st, w_x, b_ctx, b_st, b_x, pgen);
  // 13. pv_h = [s_t,ctx] @ W_pv1 + b_pv1
  wmma_gemm<64, 64, false, false><<<dim3(H_ / 64, 1), 256, 0, stream>>>(
      pv_in, 2 * H_, W_pv1, H_, pv_h, H_, b_pv1, nullptr, B_, H_, 2 * H_);
  // 14. logits = pv_h @ W_pv2 + b_pv2  (131MB weight read exactly once)
  wmma_gemm<64, 64, false, false><<<dim3(V_ / 64, 1), 256, 0, stream>>>(
      pv_h, H_, W_pv2, V_, logits, V_, b_pv2, nullptr, B_, V_, H_);
  // 15. softmax-V stats
  k_softmax_V_stats<<<B_, 1024, 0, stream>>>(logits, vmax, vsum);
  // 16. scatter-add copy distribution
  k_scatter<<<(B_ * L_ + 255) / 256, 256, 0, stream>>>(src, a_t, copyd);
  // 17. final blend
  k_final<<<(B_ * V_ + 255) / 256, 256, 0, stream>>>(logits, copyd, pgen, vmax, vsum, P_final);
}
